// GAT_71622874628670
// MI455X (gfx1250) — matrix operand order repaired and checked by reading, not compiled, not run on hardware
//
#include <hip/hip_runtime.h>
#include <hip/hip_bf16.h>
#include <math.h>

// ---------------------------------------------------------------------------
// GAT (2-layer, PyG-style) for gfx1250.
// Dense GEMMs use v_wmma_f32_16x16x32_f16 (f16 inputs, f32 accumulate).
// Edge softmax/aggregation uses float atomic max (ordered-uint trick) +
// atomicAdd scatter. All kernels on `stream`; scratch in d_ws.
// ---------------------------------------------------------------------------

typedef __attribute__((ext_vector_type(16))) _Float16 v16h;
typedef __attribute__((ext_vector_type(8)))  float    v8f;

#define C_IN      512
#define NEG_SLOPE 0.2f
#define ELU_ALPHA 0.2f

// ---- float <-> order-preserving uint (for atomic float max) ----
__device__ __forceinline__ unsigned f2ord(float f) {
  unsigned u = __float_as_uint(f);
  return (u & 0x80000000u) ? ~u : (u | 0x80000000u);
}
__device__ __forceinline__ float ord2f(unsigned u) {
  return (u & 0x80000000u) ? __uint_as_float(u ^ 0x80000000u)
                           : __uint_as_float(~u);
}

// ---------------------------------------------------------------------------
// Init kernels
// ---------------------------------------------------------------------------
__global__ void k_fill_f32(float* p, float v, int n) {
  int i = blockIdx.x * blockDim.x + threadIdx.x;
  if (i < n) p[i] = v;
}
__global__ void k_fill_u32(unsigned* p, unsigned v, int n) {
  int i = blockIdx.x * blockDim.x + threadIdx.x;
  if (i < n) p[i] = v;
}
__global__ void k_init_out_bias(float* out, const float* __restrict__ b,
                                int N, int C) {
  int i = blockIdx.x * blockDim.x + threadIdx.x;
  if (i < N * C) out[i] = b[i % C];
}

// ---------------------------------------------------------------------------
// WMMA GEMM: out[N,64] = A[N,K] @ W[K,64]  (f16 multiply, f32 accumulate)
// Block: 256 threads = 8 waves, 128 rows per block; each wave: 16 rows x 64 cols.
// W staged in LDS transposed (col-major) so B fragments are contiguous
// (ds_load_b128 pairs). A stream prefetched one K-step ahead.
// ---------------------------------------------------------------------------
template <int K>
__global__ __launch_bounds__(256) void k_gemm_wmma(
    const float* __restrict__ A, const float* __restrict__ W,
    float* __restrict__ out, int N) {
  __shared__ _Float16 sW[64 * K];  // sW[c*K + k] = (f16) W[k*64 + c]
  for (int i = threadIdx.x; i < 64 * K; i += 256) {
    int k = i >> 6, c = i & 63;
    sW[c * K + k] = (_Float16)W[i];
  }
  __syncthreads();

  const int lane = threadIdx.x & 31;
  const int wave = threadIdx.x >> 5;
  const int row0 = blockIdx.x * 128 + wave * 16;
  if (row0 >= N) return;  // wave-uniform exit: EXEC stays all-ones for WMMA

  int r = row0 + (lane & 15);
  if (r >= N) r = N - 1;  // clamp tail rows (stores are guarded below)

  const int hb   = lane >> 4;   // lane half: 0 or 1
  const int colb = lane & 15;

  v8f acc[4] = {v8f{}, v8f{}, v8f{}, v8f{}};

  const float* rowp = A + (size_t)r * K;

  for (int k0 = 0; k0 < K; k0 += 32) {
    // A fragment (16x32 f16): lane holds row (lane&15),
    // K = {kb..kb+7} in halves 0..7 and {kb+16..kb+23} in halves 8..15
    const int kbA = k0 + hb * 8;
    const float* xp = rowp + kbA;
    if (k0 + 32 < K) {
      // keep the HBM A-stream one K-step ahead (global_prefetch_b8)
      __builtin_prefetch(xp + 32, 0, 1);
    }
    v16h a;
#pragma unroll
    for (int i = 0; i < 8; ++i) {
      a[i]     = (_Float16)xp[i];
      a[i + 8] = (_Float16)xp[i + 16];
    }
    // B fragments (32x16 f16): lane holds col (lane&15),
    // K = {kb..kb+7} in halves 0..7 and {kb+16..kb+23} in halves 8..15
    const int kbB = k0 + hb * 8;
#pragma unroll
    for (int ct = 0; ct < 4; ++ct) {
      const _Float16* wp = &sW[(ct * 16 + colb) * K + kbB];
      v16h b;
#pragma unroll
      for (int j = 0; j < 8; ++j) {
        b[j]     = wp[j];
        b[j + 8] = wp[j + 16];
      }
      acc[ct] = __builtin_amdgcn_wmma_f32_16x16x32_f16(
          false, a, false, b, (short)0, acc[ct], false, false);
    }
  }

  // C/D layout: acc[ct][j] -> row = row0 + j + hb*8, col = ct*16 + (lane&15)
  const int orow0 = row0 + hb * 8;
  float* obase = out + (size_t)orow0 * 64 + colb;
  if (row0 + 16 <= N) {
    // fast path (all blocks except the last): unguarded stores, no exec churn
#pragma unroll
    for (int ct = 0; ct < 4; ++ct) {
#pragma unroll
      for (int j = 0; j < 8; ++j) {
        obase[(size_t)j * 64 + ct * 16] = acc[ct][j];
      }
    }
  } else {
#pragma unroll
    for (int ct = 0; ct < 4; ++ct) {
#pragma unroll
      for (int j = 0; j < 8; ++j) {
        if (orow0 + j < N) obase[(size_t)j * 64 + ct * 16] = acc[ct][j];
      }
    }
  }
}

// ---------------------------------------------------------------------------
// a_src[n,h] = sum_f h[n,h,f] * att_src[h,f]   (and same for a_dst)
// ---------------------------------------------------------------------------
__global__ void k_att(const float* __restrict__ h,
                      const float* __restrict__ att_s,
                      const float* __restrict__ att_d,
                      float* __restrict__ as, float* __restrict__ ad,
                      int N, int H, int F) {
  int i = blockIdx.x * blockDim.x + threadIdx.x;
  if (i >= N * H) return;
  int n = i / H, hh = i % H;
  const float* hp = h + (size_t)n * H * F + (size_t)hh * F;
  const float* sp = att_s + hh * F;
  const float* dp = att_d + hh * F;
  float accs = 0.f, accd = 0.f;
  for (int f = 0; f < F; ++f) {
    float v = hp[f];
    accs += v * sp[f];
    accd += v * dp[f];
  }
  as[i] = accs;
  ad[i] = accd;
}

// ---------------------------------------------------------------------------
// Per-edge passes. Edges 0..E-1 come from edge_index; E..E+N-1 are self loops.
// ---------------------------------------------------------------------------
__device__ __forceinline__ float edge_logit(const int* srcI, const int* dstI,
                                            const float* as, const float* ad,
                                            long long eid, int E, int H, int hh,
                                            int* s_out, int* d_out) {
  int s = (eid < E) ? srcI[eid] : (int)(eid - E);
  int d = (eid < E) ? dstI[eid] : (int)(eid - E);
  *s_out = s;
  *d_out = d;
  float e = as[(size_t)s * H + hh] + ad[(size_t)d * H + hh];
  return (e > 0.f) ? e : NEG_SLOPE * e;
}

__global__ void k_edge_max(const int* __restrict__ srcI,
                           const int* __restrict__ dstI,
                           const float* __restrict__ as,
                           const float* __restrict__ ad,
                           unsigned* __restrict__ menc,
                           long long EL, int E, int H) {
  long long i = (long long)blockIdx.x * blockDim.x + threadIdx.x;
  if (i >= EL * H) return;
  long long eid = i / H;
  int hh = (int)(i % H);
  int s, d;
  float e = edge_logit(srcI, dstI, as, ad, eid, E, H, hh, &s, &d);
  atomicMax(&menc[(size_t)d * H + hh], f2ord(e));
}

__global__ void k_max_fix(unsigned* __restrict__ menc, float* __restrict__ m,
                          int n) {
  int i = blockIdx.x * blockDim.x + threadIdx.x;
  if (i >= n) return;
  float v = ord2f(menc[i]);
  if (!isfinite(v)) v = 0.f;
  m[i] = v;  // m aliases menc: in-place decode, one thread per element
}

__global__ void k_edge_expsum(const int* __restrict__ srcI,
                              const int* __restrict__ dstI,
                              const float* __restrict__ as,
                              const float* __restrict__ ad,
                              const float* __restrict__ m,
                              float* __restrict__ denom,
                              long long EL, int E, int H) {
  long long i = (long long)blockIdx.x * blockDim.x + threadIdx.x;
  if (i >= EL * H) return;
  long long eid = i / H;
  int hh = (int)(i % H);
  int s, d;
  float e = edge_logit(srcI, dstI, as, ad, eid, E, H, hh, &s, &d);
  float ex = __expf(e - m[(size_t)d * H + hh]);
  atomicAdd(&denom[(size_t)d * H + hh], ex);
}

__global__ void k_edge_agg(const int* __restrict__ srcI,
                           const int* __restrict__ dstI,
                           const float* __restrict__ as,
                           const float* __restrict__ ad,
                           const float* __restrict__ m,
                           const float* __restrict__ denom,
                           const float* __restrict__ hfeat,
                           float* __restrict__ out,
                           long long EL, int E, int H, int F) {
  long long i = (long long)blockIdx.x * blockDim.x + threadIdx.x;
  if (i >= EL * H) return;
  long long eid = i / H;
  int hh = (int)(i % H);
  int s, d;
  float e = edge_logit(srcI, dstI, as, ad, eid, E, H, hh, &s, &d);
  size_t dH = (size_t)d * H + hh;
  float alpha = __expf(e - m[dH]) / (denom[dH] + 1e-16f);
  const float* hp = hfeat + ((size_t)s * H + hh) * F;
  float* op = out + ((size_t)d * H + hh) * F;
  for (int f = 0; f < F; ++f) atomicAdd(&op[f], hp[f] * alpha);
}

// out = elu(in + bias)  (per element, bias broadcast over C)
__global__ void k_bias_elu(const float* __restrict__ in,
                           const float* __restrict__ b,
                           float* __restrict__ out, int n, int C) {
  int i = blockIdx.x * blockDim.x + threadIdx.x;
  if (i >= n) return;
  float v = in[i] + b[i % C];
  out[i] = (v > 0.f) ? v : ELU_ALPHA * (__expf(v) - 1.f);
}

// ---------------------------------------------------------------------------
// Launcher
// ---------------------------------------------------------------------------
extern "C" void kernel_launch(void* const* d_in, const int* in_sizes, int n_in,
                              void* d_out, int out_size, void* d_ws,
                              size_t ws_size, hipStream_t stream) {
  const float* x      = (const float*)d_in[0];
  const int*   ei     = (const int*)d_in[1];
  const float* W1     = (const float*)d_in[2];
  const float* att_s1 = (const float*)d_in[3];
  const float* att_d1 = (const float*)d_in[4];
  const float* b1     = (const float*)d_in[5];
  const float* W2     = (const float*)d_in[6];
  const float* att_s2 = (const float*)d_in[7];
  const float* att_d2 = (const float*)d_in[8];
  const float* b2     = (const float*)d_in[9];

  const int N = in_sizes[0] / C_IN;
  const int E = in_sizes[1] / 2;
  const long long EL = (long long)E + N;  // edges + self loops
  const int* srcI = ei;
  const int* dstI = ei + E;

  // ---- workspace layout (f32 elements), ~59 MB total ----
  float* ws = (float*)d_ws;
  size_t off = 0;
  float* h1  = ws + off; off += (size_t)N * 64;  // conv1 linear out [N,8,8]
  float* as1 = ws + off; off += (size_t)N * 8;
  float* ad1 = ws + off; off += (size_t)N * 8;
  float* m1  = ws + off; off += (size_t)N * 8;   // uint-encoded during max pass
  float* dn1 = ws + off; off += (size_t)N * 8;
  float* o1  = ws + off; off += (size_t)N * 64;  // conv1 aggregated
  float* h2  = ws + off; off += (size_t)N * 64;  // elu(o1 + b1)
  float* t2  = ws + off; off += (size_t)N * 64;  // conv2 linear out
  float* as2 = ws + off; off += (size_t)N;
  float* ad2 = ws + off; off += (size_t)N;
  float* m2  = ws + off; off += (size_t)N;
  float* dn2 = ws + off; off += (size_t)N;
  (void)ws_size;

  const unsigned NEG_INF_ORD = 0x007FFFFFu;  // f2ord(-inf)
  auto nb = [](long long n, int b) { return (unsigned)((n + b - 1) / b); };
  const int B = 256;
  float* out = (float*)d_out;

  // ---- init ----
  k_fill_u32<<<nb((long long)N * 8, B), B, 0, stream>>>((unsigned*)m1,
                                                        NEG_INF_ORD, N * 8);
  k_fill_f32<<<nb((long long)N * 8, B), B, 0, stream>>>(dn1, 0.f, N * 8);
  k_fill_f32<<<nb((long long)N * 64, B), B, 0, stream>>>(o1, 0.f, N * 64);
  k_fill_u32<<<nb(N, B), B, 0, stream>>>((unsigned*)m2, NEG_INF_ORD, N);
  k_fill_f32<<<nb(N, B), B, 0, stream>>>(dn2, 0.f, N);
  k_init_out_bias<<<nb((long long)N * 64, B), B, 0, stream>>>(out, b2, N, 64);

  // ---- layer 1: GEMM (WMMA) + attention + softmax + aggregate ----
  k_gemm_wmma<512><<<nb(N, 128), 256, 0, stream>>>(x, W1, h1, N);
  k_att<<<nb((long long)N * 8, B), B, 0, stream>>>(h1, att_s1, att_d1, as1, ad1,
                                                   N, 8, 8);
  k_edge_max<<<nb(EL * 8, B), B, 0, stream>>>(srcI, dstI, as1, ad1,
                                              (unsigned*)m1, EL, E, 8);
  k_max_fix<<<nb((long long)N * 8, B), B, 0, stream>>>((unsigned*)m1, m1,
                                                       N * 8);
  k_edge_expsum<<<nb(EL * 8, B), B, 0, stream>>>(srcI, dstI, as1, ad1, m1, dn1,
                                                 EL, E, 8);
  k_edge_agg<<<nb(EL * 8, B), B, 0, stream>>>(srcI, dstI, as1, ad1, m1, dn1, h1,
                                              o1, EL, E, 8, 8);
  k_bias_elu<<<nb((long long)N * 64, B), B, 0, stream>>>(o1, b1, h2, N * 64,
                                                         64);

  // ---- layer 2: GEMM (WMMA) + attention + softmax + aggregate into d_out ----
  k_gemm_wmma<64><<<nb(N, 128), 256, 0, stream>>>(h2, W2, t2, N);
  k_att<<<nb(N, B), B, 0, stream>>>(t2, att_s2, att_d2, as2, ad2, N, 1, 64);
  k_edge_max<<<nb(EL, B), B, 0, stream>>>(srcI, dstI, as2, ad2, (unsigned*)m2,
                                          EL, E, 1);
  k_max_fix<<<nb(N, B), B, 0, stream>>>((unsigned*)m2, m2, N);
  k_edge_expsum<<<nb(EL, B), B, 0, stream>>>(srcI, dstI, as2, ad2, m2, dn2, EL,
                                             E, 1);
  k_edge_agg<<<nb(EL, B), B, 0, stream>>>(srcI, dstI, as2, ad2, m2, dn2, t2,
                                          out, EL, E, 1, 64);
}
